// EncoderLayer_90683939488257
// MI455X (gfx1250) — matrix operand order repaired and checked by reading, not compiled, not run on hardware
//
#include <hip/hip_runtime.h>
#include <hip/hip_bf16.h>

#define EMBED 1024
#define FFDIM 4096
#define SEQ   2048
#define BATCH 2
#define NHEAD 16
#define HDIM  64
#define ROWS  (BATCH * SEQ)   // 4096 token rows

typedef __bf16 bf16;
typedef __attribute__((ext_vector_type(16))) __bf16 v16bf;
typedef __attribute__((ext_vector_type(8)))  __bf16 v8bf;
typedef __attribute__((ext_vector_type(8)))  float  v8f;

union V16 { v16bf v; struct { v8bf lo; v8bf hi; } s; };

__device__ __forceinline__ v8f wmma_bf16(v16bf a, v16bf b, v8f c) {
  // D = A(16x32 bf16) * B(32x16 bf16) + C(16x16 f32)
  return __builtin_amdgcn_wmma_f32_16x16x32_bf16(false, a, false, b, (short)0, c,
                                                 false, false);
}

// Async copy 16 bytes global -> LDS, tracked by ASYNCcnt (CDNA5 path).
// VDST operand holds the LDS byte offset; generic pointers to __shared__
// carry the LDS offset in their low 32 bits.
__device__ __forceinline__ void async_ld16(const bf16* g, bf16* l) {
  unsigned lds_off = (unsigned)(uintptr_t)l;
  unsigned long long ga = (unsigned long long)(uintptr_t)g;
  asm volatile("global_load_async_to_lds_b128 %0, %1, off"
               :: "v"(lds_off), "v"(ga) : "memory");
}

// ---------------------------------------------------------------- convert
__global__ void __launch_bounds__(256) cvt_kernel(const float* __restrict__ src,
                                                  bf16* __restrict__ dst, int n) {
  int i = blockIdx.x * 256 + threadIdx.x;
  if (i < n) dst[i] = (bf16)src[i];
}

// w[k][n] (fp32) -> wT[n][k] (bf16): makes WMMA B-fragments a single
// contiguous 32B load per lane (K contiguous per column).
__global__ void __launch_bounds__(256) wtrans_kernel(const float* __restrict__ w,
                                                     bf16* __restrict__ wT,
                                                     int K, int N) {
  int i = blockIdx.x * 256 + threadIdx.x;
  if (i >= K * N) return;
  int k = i / N, n = i - k * N;          // reads coalesced along n
  wT[(size_t)n * K + k] = (bf16)w[i];
}

// ---------------------------------------------------------------- attention
// Flash attention, Q=K=V = x[b,:,h*64:(h+1)*64] in bf16.
// 1 block = 4 waves; wave w owns 16 query rows. Key loop in tiles of 64 keys,
// double-buffered via global_load_async_to_lds_b128 (ASYNCcnt).
__global__ void __launch_bounds__(128) attn_kernel(const bf16* __restrict__ xb,
                                                   bf16* __restrict__ outb) {
  __shared__ bf16 Ks[2][64 * 64];   // K/V tile row-major [key][d], double buffer
  __shared__ bf16 Vt[64 * 64];      // V tile transposed: [d][key]
  __shared__ bf16 Pt[4][16 * 64];   // per-wave probability tile [q][key]
  const int tid = threadIdx.x, lane = tid & 31, w = tid >> 5;
  const int half = lane >> 4, ln = lane & 15;
  int bid = blockIdx.x;
  const int qt = bid & 31; bid >>= 5;
  const int h  = bid & 15; bid >>= 4;
  const int b  = bid;
  const int q0 = qt * 64 + w * 16;
  const bf16* xbh = xb + (size_t)b * SEQ * EMBED + h * HDIM;

  // per-thread slice of the 64x64 tile: 4 async b128 issues
  const int rlane = tid >> 3;        // 0..15
  const int c8 = (tid & 7) * 8;      // 0..56

  // Q A-fragments (K = head dim 64 -> two 32-chunks). Two 16B runs per lane.
  V16 aQ[2];
  {
    const bf16* qrow = xbh + (size_t)(q0 + ln) * EMBED;
    for (int c = 0; c < 2; ++c) {
      aQ[c].s.lo = *(const v8bf*)(qrow + 32 * c + 16 * half);
      aQ[c].s.hi = *(const v8bf*)(qrow + 32 * c + 16 * half + 8);
    }
  }

  float m[8], l[8];
  v8f acc[4];
  for (int r = 0; r < 8; ++r) { m[r] = -__builtin_inff(); l[r] = 0.f; }
  for (int n = 0; n < 4; ++n) for (int r = 0; r < 8; ++r) acc[n][r] = 0.f;

  // prologue: issue async load of tile 0
#pragma unroll
  for (int j = 0; j < 4; ++j) {
    int row = j * 16 + rlane;
    async_ld16(xbh + (size_t)row * EMBED + c8, &Ks[0][row * 64 + c8]);
  }

  for (int kt = 0; kt < SEQ; kt += 64) {
    const int buf = (kt >> 6) & 1;
    bf16* ks = Ks[buf];
    // only tile kt is in flight here; wait for it, then barrier. The barrier
    // also guarantees everyone is done reading Ks[buf^1] (tile kt-64).
    asm volatile("s_wait_asynccnt 0x0" ::: "memory");
    __syncthreads();
    if (kt + 64 < SEQ) {             // overlap next tile fetch with compute
#pragma unroll
      for (int j = 0; j < 4; ++j) {
        int row = j * 16 + rlane;
        async_ld16(xbh + (size_t)(kt + 64 + row) * EMBED + c8,
                   &Ks[buf ^ 1][row * 64 + c8]);
      }
    }
    // Transpose V tile (LDS -> regs -> LDS)
    for (int i = tid; i < 512; i += 128) {
      int key = i >> 3, d0 = (i & 7) * 8;
      v8bf v = *(const v8bf*)(ks + key * 64 + d0);
      for (int j = 0; j < 8; ++j) Vt[(d0 + j) * 64 + key] = v[j];
    }
    __syncthreads();

    // S = Q K^T: B-fragment = 16 contiguous head-dim values of one key (LDS).
    // Preload all 8 B-fragments, then run the WMMA chain.
    v16bf bk[4][2];
#pragma unroll
    for (int n = 0; n < 4; ++n)
#pragma unroll
      for (int c = 0; c < 2; ++c)
        bk[n][c] = *(const v16bf*)(ks + (n * 16 + ln) * 64 + 32 * c + 16 * half);
    v8f s[4];
#pragma unroll
    for (int n = 0; n < 4; ++n) {
      for (int r = 0; r < 8; ++r) s[n][r] = 0.f;
      s[n] = wmma_bf16(aQ[0].v, bk[n][0], s[n]);
      s[n] = wmma_bf16(aQ[1].v, bk[n][1], s[n]);
    }
    for (int n = 0; n < 4; ++n)
      for (int r = 0; r < 8; ++r) s[n][r] *= 0.125f;   // 1/sqrt(64)

    // Online softmax. Row M=r+8*half lives across the 16 lanes of each half:
    // xor masks 1..8 reduce within a half.
    float rm[8], rs[8];
    for (int r = 0; r < 8; ++r) {
      float t = fmaxf(fmaxf(s[0][r], s[1][r]), fmaxf(s[2][r], s[3][r]));
      for (int mask = 1; mask < 16; mask <<= 1) t = fmaxf(t, __shfl_xor(t, mask, 32));
      rm[r] = t;
    }
    for (int r = 0; r < 8; ++r) {
      float mn = fmaxf(m[r], rm[r]);
      float al = __expf(m[r] - mn);
      m[r] = mn;
      l[r] *= al;
      for (int n = 0; n < 4; ++n) acc[n][r] *= al;
      float sum = 0.f;
      for (int n = 0; n < 4; ++n) { float p = __expf(s[n][r] - mn); s[n][r] = p; sum += p; }
      rs[r] = sum;
    }
    for (int r = 0; r < 8; ++r) {
      for (int mask = 1; mask < 16; mask <<= 1) rs[r] += __shfl_xor(rs[r], mask, 32);
      l[r] += rs[r];
    }

    // C-layout -> A-layout for P via per-wave LDS tile
    bf16* pt = Pt[w];
    for (int n = 0; n < 4; ++n)
      for (int r = 0; r < 8; ++r)
        pt[(r + 8 * half) * 64 + n * 16 + ln] = (bf16)s[n][r];
    asm volatile("s_wait_dscnt 0x0" ::: "memory");   // wave-internal LDS RAW

    // O += P * V  (B-fragment = 16 contiguous keys of one d from Vt)
#pragma unroll
    for (int c = 0; c < 2; ++c) {
      V16 aP;
      aP.s.lo = *(const v8bf*)(pt + ln * 64 + 32 * c + 16 * half);
      aP.s.hi = *(const v8bf*)(pt + ln * 64 + 32 * c + 16 * half + 8);
      v16bf bv[4];
#pragma unroll
      for (int n = 0; n < 4; ++n)
        bv[n] = *(const v16bf*)(&Vt[(n * 16 + ln) * 64 + 32 * c + 16 * half]);
#pragma unroll
      for (int n = 0; n < 4; ++n)
        acc[n] = wmma_bf16(aP.v, bv[n], acc[n]);
    }
  }

  for (int n = 0; n < 4; ++n)
    for (int r = 0; r < 8; ++r) {
      float o = acc[n][r] / l[r];
      outb[(size_t)b * SEQ * EMBED + (size_t)(q0 + r + 8 * half) * EMBED +
           h * HDIM + n * 16 + ln] = (bf16)o;
    }
}

// -------------------------------------------- GEMM + bias + residual + LayerNorm
// Block = 16 rows x 1024 cols (8 waves x 128 cols) so LN reduction is in-block.
__global__ void __launch_bounds__(256) gemm_bias_res_ln_kernel(
    const bf16* __restrict__ A, const bf16* __restrict__ WT,
    const float* __restrict__ bias, const float* __restrict__ resid,
    const float* __restrict__ gamma, const float* __restrict__ beta,
    float* __restrict__ outF, bf16* __restrict__ outB, int Kdim) {
  __shared__ float s_sum[8][16];
  __shared__ float s_ssq[8][16];
  const int tid = threadIdx.x, lane = tid & 31, w = tid >> 5;
  const int half = lane >> 4, ln = lane & 15;
  const int row0 = blockIdx.x * 16;
  const int col0 = w * 128;

  v8f acc[8];
  for (int n = 0; n < 8; ++n) for (int r = 0; r < 8; ++r) acc[n][r] = 0.f;

  const bf16* arow0 = A + (size_t)(row0 + ln) * Kdim;
  const bf16* bbase0 = WT + (size_t)(col0 + ln) * Kdim + 16 * half;
  for (int kc = 0; kc < Kdim / 32; ++kc) {
    const bf16* arow = arow0 + kc * 32;
    __builtin_prefetch(arow + 64, 0, 0);        // global_prefetch_b8, kc+2 ahead
    V16 a;
    a.s.lo = *(const v8bf*)(arow + 16 * half);
    a.s.hi = *(const v8bf*)(arow + 16 * half + 8);
    // batch the 8 B-fragment loads so they clause & overlap, then WMMA chain
    v16bf bfr[8];
    const bf16* bbase = bbase0 + kc * 32;
#pragma unroll
    for (int n = 0; n < 8; ++n)
      bfr[n] = *(const v16bf*)(bbase + (size_t)n * 16 * Kdim);
#pragma unroll
    for (int n = 0; n < 8; ++n)
      acc[n] = wmma_bf16(a.v, bfr[n], acc[n]);
  }

  float vals[8][8], psum[8], pssq[8];
  for (int r = 0; r < 8; ++r) { psum[r] = 0.f; pssq[r] = 0.f; }
  for (int n = 0; n < 8; ++n)
    for (int r = 0; r < 8; ++r) {
      int col = col0 + n * 16 + ln;
      int row = row0 + r + 8 * half;
      float v = acc[n][r] + bias[col] + resid[(size_t)row * EMBED + col];
      vals[n][r] = v; psum[r] += v; pssq[r] += v * v;
    }
  for (int r = 0; r < 8; ++r)
    for (int mask = 1; mask < 16; mask <<= 1) {
      psum[r] += __shfl_xor(psum[r], mask, 32);
      pssq[r] += __shfl_xor(pssq[r], mask, 32);
    }
  if (ln == 0)  // lanes 0 and 16: rows r and r+8 respectively
    for (int r = 0; r < 8; ++r) {
      s_sum[w][r + 8 * half] = psum[r];
      s_ssq[w][r + 8 * half] = pssq[r];
    }
  __syncthreads();
  float mu[8], rstd[8];
  for (int r = 0; r < 8; ++r) {
    int row = r + 8 * half;
    float s = 0.f, q = 0.f;
    for (int ww = 0; ww < 8; ++ww) { s += s_sum[ww][row]; q += s_ssq[ww][row]; }
    float mm = s * (1.f / EMBED);
    float var = q * (1.f / EMBED) - mm * mm;
    mu[r] = mm; rstd[r] = rsqrtf(var + 1e-5f);
  }
  for (int n = 0; n < 8; ++n)
    for (int r = 0; r < 8; ++r) {
      int col = col0 + n * 16 + ln;
      size_t row = row0 + r + 8 * half;
      float y = (vals[n][r] - mu[r]) * rstd[r] * gamma[col] + beta[col];
      outF[row * EMBED + col] = y;
      if (outB) outB[row * EMBED + col] = (bf16)y;
    }
}

// -------------------------------------------- GEMM + bias + ReLU (ff1)
__global__ void __launch_bounds__(256) gemm_bias_relu_kernel(
    const bf16* __restrict__ A, const bf16* __restrict__ WT,
    const float* __restrict__ bias, bf16* __restrict__ H) {
  const int tid = threadIdx.x, lane = tid & 31, w = tid >> 5;
  const int half = lane >> 4, ln = lane & 15;
  const int row0 = blockIdx.x * 16;
  const int col0 = blockIdx.y * 1024 + w * 128;
  const int Kdim = EMBED;

  v8f acc[8];
  for (int n = 0; n < 8; ++n) for (int r = 0; r < 8; ++r) acc[n][r] = 0.f;

  const bf16* arow0 = A + (size_t)(row0 + ln) * Kdim;
  const bf16* bbase0 = WT + (size_t)(col0 + ln) * Kdim + 16 * half;
  for (int kc = 0; kc < Kdim / 32; ++kc) {
    const bf16* arow = arow0 + kc * 32;
    __builtin_prefetch(arow + 64, 0, 0);
    V16 a;
    a.s.lo = *(const v8bf*)(arow + 16 * half);
    a.s.hi = *(const v8bf*)(arow + 16 * half + 8);
    v16bf bfr[8];
    const bf16* bbase = bbase0 + kc * 32;
#pragma unroll
    for (int n = 0; n < 8; ++n)
      bfr[n] = *(const v16bf*)(bbase + (size_t)n * 16 * Kdim);
#pragma unroll
    for (int n = 0; n < 8; ++n)
      acc[n] = wmma_bf16(a.v, bfr[n], acc[n]);
  }
  for (int n = 0; n < 8; ++n)
    for (int r = 0; r < 8; ++r) {
      int col = col0 + n * 16 + ln;
      size_t row = row0 + r + 8 * half;
      float y = fmaxf(acc[n][r] + bias[col], 0.f);
      H[row * FFDIM + col] = (bf16)y;
    }
}

// ---------------------------------------------------------------- launcher
extern "C" void kernel_launch(void* const* d_in, const int* in_sizes, int n_in,
                              void* d_out, int out_size, void* d_ws, size_t ws_size,
                              hipStream_t stream) {
  const float* x     = (const float*)d_in[0];
  const float* w_fc  = (const float*)d_in[1];
  const float* b_fc  = (const float*)d_in[2];
  const float* g1    = (const float*)d_in[3];
  const float* be1   = (const float*)d_in[4];
  const float* w_ff1 = (const float*)d_in[5];
  const float* b_ff1 = (const float*)d_in[6];
  const float* w_ff2 = (const float*)d_in[7];
  const float* b_ff2 = (const float*)d_in[8];
  const float* g2    = (const float*)d_in[9];
  const float* be2   = (const float*)d_in[10];
  float* out = (float*)d_out;

  char* ws = (char*)d_ws;
  size_t off = 0;
  auto give = [&](size_t bytes) -> char* {
    char* p = ws + off;
    off = (off + bytes + 255) & ~(size_t)255;
    return p;
  };
  // total ~91 MB of workspace
  bf16*  xb   = (bf16*)give((size_t)ROWS * EMBED * 2);
  bf16*  outb = (bf16*)give((size_t)ROWS * EMBED * 2);
  bf16*  x1b  = (bf16*)give((size_t)ROWS * EMBED * 2);
  float* x1f  = (float*)give((size_t)ROWS * EMBED * 4);
  bf16*  hb   = (bf16*)give((size_t)ROWS * FFDIM * 2);
  bf16*  wfT  = (bf16*)give((size_t)EMBED * EMBED * 2);
  bf16*  w1T  = (bf16*)give((size_t)EMBED * FFDIM * 2);
  bf16*  w2T  = (bf16*)give((size_t)FFDIM * EMBED * 2);

  cvt_kernel<<<(ROWS * EMBED + 255) / 256, 256, 0, stream>>>(x, xb, ROWS * EMBED);
  wtrans_kernel<<<(EMBED * EMBED + 255) / 256, 256, 0, stream>>>(w_fc, wfT, EMBED, EMBED);
  wtrans_kernel<<<(EMBED * FFDIM + 255) / 256, 256, 0, stream>>>(w_ff1, w1T, EMBED, FFDIM);
  wtrans_kernel<<<(FFDIM * EMBED + 255) / 256, 256, 0, stream>>>(w_ff2, w2T, FFDIM, EMBED);

  attn_kernel<<<BATCH * NHEAD * (SEQ / 64), 128, 0, stream>>>(xb, outb);

  // attn_out @ w_fc_out + b ; x1 = LN(x + .)
  gemm_bias_res_ln_kernel<<<ROWS / 16, 256, 0, stream>>>(
      outb, wfT, b_fc, x, g1, be1, x1f, x1b, EMBED);
  // h = relu(x1 @ w_ff1 + b1)
  gemm_bias_relu_kernel<<<dim3(ROWS / 16, FFDIM / 1024), 256, 0, stream>>>(
      x1b, w1T, b_ff1, hb);
  // out = LN(x1 + h @ w_ff2 + b2)
  gemm_bias_res_ln_kernel<<<ROWS / 16, 256, 0, stream>>>(
      hb, w2T, b_ff2, x1f, g2, be2, out, (bf16*)nullptr, FFDIM);
}
